// CopyDecoder_62380105008067
// MI455X (gfx1250) — matrix operand order repaired and checked by reading, not compiled, not run on hardware
//
#include <hip/hip_runtime.h>
#include <hip/hip_bf16.h>

#define B_    32
#define S_    1024
#define V_    32000
#define E_    256
#define H_    512
#define OOV_  12
#define D2_   1124      // 2H + 100
#define IN_   1380      // E + D2
#define KPAD_ 1152      // D2 padded to multiple of 32
#define VO_   32012     // V + OOV

typedef __attribute__((ext_vector_type(16))) __bf16 v16bf;
typedef __attribute__((ext_vector_type(8)))  __bf16 v8bf;
typedef __attribute__((ext_vector_type(8)))  float  v8f;

static __device__ __forceinline__ __bf16 f2b(float x) { return (__bf16)x; }

static __device__ __forceinline__ v8f v8f_zero() {
  v8f z = {0.f, 0.f, 0.f, 0.f, 0.f, 0.f, 0.f, 0.f};
  return z;
}

static __device__ __forceinline__ v8f wmma_bf16(v16bf a, v16bf b, v8f c) {
  // (neg_a, A, neg_b, B, c_mod, C, reuse_a, reuse_b)
  return __builtin_amdgcn_wmma_f32_16x16x32_bf16(false, a, false, b, (short)0, c,
                                                 false, false);
}

// A-matrix fragment (16x32 bf16): lane holds row m = lane&15.
// Layout: lanes 0-15 hold K = 0..15; lanes 16-31 K = 16..31 (matches B fragments).
// row points at element (m, k=0) of the 32-wide k tile; 16B-aligned chunks.
static __device__ __forceinline__ v16bf load_a_frag(const __bf16* row, int lane) {
  int koff = (lane & 16) ? 16 : 0;
  v8bf lo = *(const v8bf*)(row + koff);
  v8bf hi = *(const v8bf*)(row + koff + 8);
  v16bf r;
#pragma unroll
  for (int i = 0; i < 8; ++i) { r[i] = lo[i]; r[i + 8] = hi[i]; }
  return r;
}

// B-matrix fragment (32x16 bf16): lane holds col n = lane&15.
// lanes 0-15: K = 0..15, lanes 16-31: K = 16..31 (ascending per v16 element).
// col points at W[n][k=0] of the 32-wide k tile (row-major weights, contiguous k).
static __device__ __forceinline__ v16bf load_b_frag_bf16(const __bf16* col, int lane) {
  int kb = (lane & 16) ? 16 : 0;
  v8bf x = *(const v8bf*)(col + kb);
  v8bf y = *(const v8bf*)(col + kb + 8);
  v16bf r;
#pragma unroll
  for (int i = 0; i < 8; ++i) { r[i] = x[i]; r[i + 8] = y[i]; }
  return r;
}

// Same as above but source is fp32 global (converted in-flight).
static __device__ __forceinline__ v16bf load_b_frag_f32(const float* col, int lane) {
  int kb = (lane & 16) ? 16 : 0;
  const float4* p = (const float4*)(col + kb);
  float4 f0 = p[0], f1 = p[1], f2 = p[2], f3 = p[3];
  v16bf r;
  r[0]  = f2b(f0.x); r[1]  = f2b(f0.y); r[2]  = f2b(f0.z); r[3]  = f2b(f0.w);
  r[4]  = f2b(f1.x); r[5]  = f2b(f1.y); r[6]  = f2b(f1.z); r[7]  = f2b(f1.w);
  r[8]  = f2b(f2.x); r[9]  = f2b(f2.y); r[10] = f2b(f2.z); r[11] = f2b(f2.w);
  r[12] = f2b(f3.x); r[13] = f2b(f3.y); r[14] = f2b(f3.z); r[15] = f2b(f3.w);
  return r;
}

// ---------------------------------------------------------------------------
// K1: resolve order branch + pre-convert Wc_w to padded bf16
// ---------------------------------------------------------------------------
__global__ void k_prep(const int* __restrict__ order,
                       const float* __restrict__ prev_state,
                       const float* __restrict__ weighted,
                       const float* __restrict__ encoded,
                       const float* __restrict__ binv,
                       const float* __restrict__ Ws_w,
                       const float* __restrict__ Ws_b,
                       const float* __restrict__ Wc_w,
                       float* __restrict__ hprev,
                       float* __restrict__ wgt,
                       __bf16* __restrict__ wcbf) {
  int idx = blockIdx.x * 256 + threadIdx.x;
  const int N1 = B_ * H_, N2 = B_ * D2_, N3 = H_ * KPAD_;
  int ord = order[0];
  if (idx < N1) {
    if (ord != 0) {
      hprev[idx] = prev_state[idx];
    } else {  // prev_state = enc[:, -1] @ Ws_w^T + Ws_b
      int b = idx / H_, h = idx % H_;
      const float* wr = Ws_w + (size_t)h * D2_;
      const float* er = encoded + ((size_t)b * S_ + (S_ - 1)) * (2 * H_);
      float bv = binv[b * S_ + (S_ - 1)];
      float acc = Ws_b[h];
      for (int d = 0; d < 2 * H_; ++d) acc += er[d] * wr[d];
      for (int d = 2 * H_; d < D2_; ++d) acc += bv * wr[d];
      hprev[idx] = acc;
    }
  } else if (idx < N1 + N2) {
    int i = idx - N1;
    wgt[i] = ord ? weighted[i] : 0.f;
  } else if (idx < N1 + N2 + N3) {
    int j = idx - N1 - N2;
    int h = j / KPAD_, k = j - h * KPAD_;
    float v = (k < D2_) ? Wc_w[(size_t)h * D2_ + k] : 0.f;
    wcbf[j] = f2b(v);
  }
}

// ---------------------------------------------------------------------------
// K2: GRU step (tiny GEMMs -> VALU dot products). Writes state into d_out.
// ---------------------------------------------------------------------------
__global__ void k_gru(const int* __restrict__ input_idx,
                      const float* __restrict__ embed,
                      const float* __restrict__ hprev,
                      const float* __restrict__ wgt,
                      const float* __restrict__ W_ih,
                      const float* __restrict__ b_ih,
                      const float* __restrict__ W_hh,
                      const float* __restrict__ b_hh,
                      float* __restrict__ state) {
  __shared__ __align__(16) float xs[IN_];
  __shared__ __align__(16) float hs[H_];
  int tid = threadIdx.x;
  int b = blockIdx.x >> 1;
  int hbase = (blockIdx.x & 1) * 256;
  int tok = input_idx[b];
  for (int i = tid; i < IN_; i += 256)
    xs[i] = (i < E_) ? embed[(size_t)tok * E_ + i] : wgt[b * D2_ + (i - E_)];
  for (int i = tid; i < H_; i += 256) hs[i] = hprev[b * H_ + i];
  __syncthreads();

  int hh = hbase + tid;
  const float4* xv = (const float4*)xs;
  const float4* hv = (const float4*)hs;
  const float4* w0 = (const float4*)(W_ih + (size_t)hh * IN_);
  const float4* w1 = (const float4*)(W_ih + (size_t)(H_ + hh) * IN_);
  const float4* w2 = (const float4*)(W_ih + (size_t)(2 * H_ + hh) * IN_);
  float g0 = b_ih[hh], g1 = b_ih[H_ + hh], g2 = b_ih[2 * H_ + hh];
  for (int i = 0; i < IN_ / 4; ++i) {
    float4 x = xv[i], a = w0[i], c = w1[i], d = w2[i];
    g0 += x.x * a.x + x.y * a.y + x.z * a.z + x.w * a.w;
    g1 += x.x * c.x + x.y * c.y + x.z * c.z + x.w * c.w;
    g2 += x.x * d.x + x.y * d.y + x.z * d.z + x.w * d.w;
  }
  const float4* u0 = (const float4*)(W_hh + (size_t)hh * H_);
  const float4* u1 = (const float4*)(W_hh + (size_t)(H_ + hh) * H_);
  const float4* u2 = (const float4*)(W_hh + (size_t)(2 * H_ + hh) * H_);
  float h0 = b_hh[hh], h1 = b_hh[H_ + hh], h2 = b_hh[2 * H_ + hh];
  for (int i = 0; i < H_ / 4; ++i) {
    float4 x = hv[i], a = u0[i], c = u1[i], d = u2[i];
    h0 += x.x * a.x + x.y * a.y + x.z * a.z + x.w * a.w;
    h1 += x.x * c.x + x.y * c.y + x.z * c.z + x.w * c.w;
    h2 += x.x * d.x + x.y * d.y + x.z * d.z + x.w * d.w;
  }
  float r = 1.f / (1.f + expf(-(g0 + h0)));
  float z = 1.f / (1.f + expf(-(g1 + h1)));
  float n = tanhf(g2 + r * h2);
  state[b * H_ + hh] = (1.f - z) * n + z * hs[hh];
}

// ---------------------------------------------------------------------------
// K3: score_g = state @ Wo_w^T + Wo_b  via bf16 WMMA. M=32 (2 tiles), N=2000
// tiles, K=512. 8 waves/block, one 16x16 output tile per wave.
// ---------------------------------------------------------------------------
__global__ void k_scoreg(const float* __restrict__ state,
                         const float* __restrict__ Wo_w,
                         const float* __restrict__ Wo_b,
                         float* __restrict__ sg) {
  __shared__ __align__(16) __bf16 Sst[B_ * H_];  // 32 KB
  int tid = threadIdx.x;
  for (int i = tid; i < B_ * H_; i += 256) Sst[i] = f2b(state[i]);
  __syncthreads();

  int wid = tid >> 5, lane = tid & 31;
  int task = blockIdx.x * 8 + wid;       // 0..3999
  int ntile = task >> 1, mtile = task & 1;
  int mrow = mtile * 16 + (lane & 15);
  int ncol = ntile * 16 + (lane & 15);

  v8f acc = v8f_zero();
  for (int kk = 0; kk < H_ / 32; ++kk) {
    v16bf a = load_a_frag(&Sst[mrow * H_ + kk * 32], lane);
    v16bf bm = load_b_frag_f32(Wo_w + (size_t)ncol * H_ + kk * 32, lane);
    acc = wmma_bf16(a, bm, acc);
  }
  float bias = Wo_b[ncol];
  int hi8 = (lane & 16) ? 8 : 0;
#pragma unroll
  for (int r = 0; r < 8; ++r) {
    int m = mtile * 16 + r + hi8;
    sg[(size_t)m * V_ + ncol] = acc[r] + bias;
  }
}

// ---------------------------------------------------------------------------
// K4: the big one.  For 64 rows of (b,s):
//   sc[row,:] = tanh(enc[row,:] @ Wc_w^T + Wc_b)   (bf16 WMMA, K=1124->1152)
//   score_c[row] = tanh(dot(sc[row,:], state[b,:])) + mask
// Fused epilogue; sc never hits memory.
// ---------------------------------------------------------------------------
__global__ void k_scorec(const float* __restrict__ encoded,
                         const float* __restrict__ binv,
                         const int* __restrict__ enc_idx,
                         const __bf16* __restrict__ wcbf,
                         const float* __restrict__ Wc_b,
                         const float* __restrict__ state,
                         float* __restrict__ scs) {
  __shared__ __align__(16) __bf16 As[64 * 32];  // 4 KB, restaged per k-step
  __shared__ float rsum[64];
  int tid = threadIdx.x, wid = tid >> 5, lane = tid & 31;
  int R = blockIdx.x * 64;     // S_ multiple of 64 -> single b per block
  int b = R / S_;
  int s0 = R - b * S_;
  for (int i = tid; i < 64; i += 256) rsum[i] = 0.f;

  v8f acc[4][4];
#pragma unroll
  for (int mi = 0; mi < 4; ++mi)
#pragma unroll
    for (int ni = 0; ni < 4; ++ni) acc[mi][ni] = v8f_zero();

  int ar = tid >> 2;          // staging row   0..63
  int ak = (tid & 3) * 8;     // staging col   {0,8,16,24}
  size_t rowoff = ((size_t)(b * S_ + s0 + ar)) * (2 * H_);
  float bvrow = binv[b * S_ + s0 + ar];

  for (int kk = 0; kk < KPAD_ / 32; ++kk) {
    int kbase = kk * 32;
    __syncthreads();
#pragma unroll
    for (int i = 0; i < 8; ++i) {  // stage enc (virtual concat) as bf16
      int k = kbase + ak + i;
      float v = (k < 2 * H_) ? encoded[rowoff + k] : ((k < D2_) ? bvrow : 0.f);
      As[ar * 32 + ak + i] = f2b(v);
    }
    __syncthreads();

    v16bf bm[4];
#pragma unroll
    for (int ni = 0; ni < 4; ++ni) {
      int n = (wid * 4 + ni) * 16 + (lane & 15);
      bm[ni] = load_b_frag_bf16(wcbf + (size_t)n * KPAD_ + kbase, lane);
    }
#pragma unroll
    for (int mi = 0; mi < 4; ++mi) {
      v16bf a = load_a_frag(&As[(mi * 16 + (lane & 15)) * 32], lane);
#pragma unroll
      for (int ni = 0; ni < 4; ++ni) acc[mi][ni] = wmma_bf16(a, bm[ni], acc[mi][ni]);
    }
  }

  // Epilogue: tanh(+bias) then dot with state along n, reduce across lanes.
  float stv[4], bsv[4];
#pragma unroll
  for (int ni = 0; ni < 4; ++ni) {
    int n = (wid * 4 + ni) * 16 + (lane & 15);
    stv[ni] = state[b * H_ + n];
    bsv[ni] = Wc_b[n];
  }
  int hi8 = (lane & 16) ? 8 : 0;
#pragma unroll
  for (int mi = 0; mi < 4; ++mi) {
#pragma unroll
    for (int r = 0; r < 8; ++r) {
      float ps = 0.f;
#pragma unroll
      for (int ni = 0; ni < 4; ++ni) ps += tanhf(acc[mi][ni][r] + bsv[ni]) * stv[ni];
      ps += __shfl_xor(ps, 1, 32);
      ps += __shfl_xor(ps, 2, 32);
      ps += __shfl_xor(ps, 4, 32);
      ps += __shfl_xor(ps, 8, 32);
      if ((lane & 15) == 0) atomicAdd(&rsum[mi * 16 + r + hi8], ps);
    }
  }
  __syncthreads();
  if (tid < 64) {
    int srow = s0 + tid;
    float v = tanhf(rsum[tid]);
    if (enc_idx[b * S_ + srow] == 0) v += -1000.f;
    scs[b * S_ + srow] = v;
  }
}

// ---------------------------------------------------------------------------
// K5: per-row softmax stats over [score_g | score_c] + copy-prob row sums +
// pointer-match counts.  stats[b] = {max, denom, prob_c_rowsum, match_count}
// ---------------------------------------------------------------------------
__global__ void k_rowstats(const float* __restrict__ sg,
                           const float* __restrict__ scs,
                           const int* __restrict__ enc_idx,
                           const int* __restrict__ input_idx,
                           float* __restrict__ stats) {
  __shared__ float red[256];
  int b = blockIdx.x, tid = threadIdx.x;
  float m = -1e30f;
  for (int j = tid; j < V_; j += 256) m = fmaxf(m, sg[(size_t)b * V_ + j]);
  for (int j = tid; j < S_; j += 256) m = fmaxf(m, scs[b * S_ + j]);
  red[tid] = m; __syncthreads();
  for (int o = 128; o > 0; o >>= 1) {
    if (tid < o) red[tid] = fmaxf(red[tid], red[tid + o]);
    __syncthreads();
  }
  float mx = red[0]; __syncthreads();

  float den = 0.f, pc = 0.f, cnt = 0.f;
  int tok = input_idx[b];
  for (int j = tid; j < V_; j += 256) den += expf(sg[(size_t)b * V_ + j] - mx);
  for (int j = tid; j < S_; j += 256) {
    float e = expf(scs[b * S_ + j] - mx);
    den += e; pc += e;
    cnt += (enc_idx[b * S_ + j] == tok) ? 1.f : 0.f;
  }
  red[tid] = den; __syncthreads();
  for (int o = 128; o > 0; o >>= 1) { if (tid < o) red[tid] += red[tid + o]; __syncthreads(); }
  float dt = red[0]; __syncthreads();
  red[tid] = pc; __syncthreads();
  for (int o = 128; o > 0; o >>= 1) { if (tid < o) red[tid] += red[tid + o]; __syncthreads(); }
  float pt = red[0]; __syncthreads();
  red[tid] = cnt; __syncthreads();
  for (int o = 128; o > 0; o >>= 1) { if (tid < o) red[tid] += red[tid + o]; __syncthreads(); }
  if (tid == 0) {
    stats[b * 4 + 0] = mx;
    stats[b * 4 + 1] = dt;
    stats[b * 4 + 2] = pt / dt;   // row sum of prob_c
    stats[b * 4 + 3] = red[0];    // match count
  }
}

// K6: out base = prob_g for j<V, 1e-4 for OOV slots
__global__ void k_outbase(const float* __restrict__ sg,
                          const float* __restrict__ stats,
                          float* __restrict__ out) {
  int idx = blockIdx.x * 256 + threadIdx.x;
  if (idx >= B_ * VO_) return;
  int b = idx / VO_, j = idx - b * VO_;
  float v = (j < V_) ? expf(sg[(size_t)b * V_ + j] - stats[b * 4]) / stats[b * 4 + 1]
                     : 1e-4f;
  out[idx] = v;
}

// K7: scatter-add prob_c into out at encoded_idx
__global__ void k_scatter(const float* __restrict__ scs,
                          const int* __restrict__ enc_idx,
                          const float* __restrict__ stats,
                          float* __restrict__ out) {
  int idx = blockIdx.x * 256 + threadIdx.x;  // B*S
  int b = idx / S_;
  float p = expf(scs[idx] - stats[b * 4]) / stats[b * 4 + 1];
  atomicAdd(&out[(size_t)b * VO_ + enc_idx[idx]], p);
}

// K8: weighted_out = attn @ enc, exploiting sparsity of encoded_idx==input_idx
__global__ void k_weighted(const float* __restrict__ encoded,
                           const float* __restrict__ binv,
                           const int* __restrict__ enc_idx,
                           const int* __restrict__ input_idx,
                           const float* __restrict__ scs,
                           const float* __restrict__ stats,
                           float* __restrict__ wout) {
  __shared__ float attA[S_];
  __shared__ int   list[S_];
  __shared__ int   cntS;
  __shared__ float acc[D2_];
  int b = blockIdx.x, tid = threadIdx.x;
  float tot = 0.f;
  for (int i = 0; i < B_; ++i) tot += stats[i * 4 + 2];  // global sum(prob_c)
  float c = stats[b * 4 + 3];
  float adj = (c > 1.f) ? c : 1.f;
  int tok = input_idx[b];
  float mx = stats[b * 4], den = stats[b * 4 + 1];
  for (int s = tid; s < S_; s += 256) {
    float a = 0.f;
    if (enc_idx[b * S_ + s] == tok) {
      float p = expf(scs[b * S_ + s] - mx) / den;
      a = (p / tot) / adj;
    }
    attA[s] = a;
  }
  for (int d = tid; d < D2_; d += 256) acc[d] = 0.f;
  __syncthreads();
  if (tid == 0) {  // deterministic in-place compaction (c2 <= s always)
    int c2 = 0;
    for (int s = 0; s < S_; ++s)
      if (attA[s] != 0.f) { list[c2] = s; attA[c2] = attA[s]; ++c2; }
    cntS = c2;
  }
  __syncthreads();
  int n = cntS;
  for (int e = 0; e < n; ++e) {
    int s = list[e];
    float a = attA[e];
    const float* er = encoded + ((size_t)(b * S_ + s)) * (2 * H_);
    float bv = binv[b * S_ + s];
    for (int d = tid; d < D2_; d += 256)
      acc[d] += a * ((d < 2 * H_) ? er[d] : bv);
  }
  __syncthreads();
  for (int d = tid; d < D2_; d += 256) wout[b * D2_ + d] = acc[d];
}

// ---------------------------------------------------------------------------
extern "C" void kernel_launch(void* const* d_in, const int* in_sizes, int n_in,
                              void* d_out, int out_size, void* d_ws, size_t ws_size,
                              hipStream_t stream) {
  const int*   input_idx = (const int*)d_in[0];
  const float* encoded   = (const float*)d_in[1];
  const int*   enc_idx   = (const int*)d_in[2];
  const float* prev_st   = (const float*)d_in[3];
  const float* weighted  = (const float*)d_in[4];
  const float* binv      = (const float*)d_in[5];
  const int*   order     = (const int*)d_in[6];
  const float* embed     = (const float*)d_in[7];
  const float* W_ih      = (const float*)d_in[8];
  const float* b_ih      = (const float*)d_in[9];
  const float* W_hh      = (const float*)d_in[10];
  const float* b_hh      = (const float*)d_in[11];
  const float* Wo_w      = (const float*)d_in[12];
  const float* Wo_b      = (const float*)d_in[13];
  const float* Wc_w      = (const float*)d_in[14];
  const float* Wc_b      = (const float*)d_in[15];
  const float* Ws_w      = (const float*)d_in[16];
  const float* Ws_b      = (const float*)d_in[17];

  float* out      = (float*)d_out;
  float* outState = out + (size_t)B_ * VO_;          // 32*32012
  float* outW     = outState + B_ * H_;              // + 32*512

  float* ws     = (float*)d_ws;
  float* hprev  = ws;                                // 16384 f
  float* wgt    = ws + 16384;                        // 35968 f
  float* sg     = ws + 52352;                        // 1024000 f
  float* scs    = ws + 1076352;                      // 32768 f
  float* stats  = ws + 1109120;                      // 128 f
  __bf16* wcbf  = (__bf16*)(ws + 1109248);           // 512*1152 bf16 (16B aligned)

  const int prepN = B_ * H_ + B_ * D2_ + H_ * KPAD_;
  k_prep    <<<(prepN + 255) / 256, 256, 0, stream>>>(order, prev_st, weighted,
                                                      encoded, binv, Ws_w, Ws_b,
                                                      Wc_w, hprev, wgt, wcbf);
  k_gru     <<<B_ * 2, 256, 0, stream>>>(input_idx, embed, hprev, wgt,
                                         W_ih, b_ih, W_hh, b_hh, outState);
  k_scoreg  <<<500, 256, 0, stream>>>(outState, Wo_w, Wo_b, sg);
  k_scorec  <<<(B_ * S_) / 64, 256, 0, stream>>>(encoded, binv, enc_idx,
                                                 wcbf, Wc_b, outState, scs);
  k_rowstats<<<B_, 256, 0, stream>>>(sg, scs, enc_idx, input_idx, stats);
  k_outbase <<<(B_ * VO_ + 255) / 256, 256, 0, stream>>>(sg, stats, out);
  k_scatter <<<(B_ * S_) / 256, 256, 0, stream>>>(scs, enc_idx, stats, out);
  k_weighted<<<B_, 256, 0, stream>>>(encoded, binv, enc_idx, input_idx,
                                     scs, stats, outW);
}
